// MultiHeadAttention_39213051412700
// MI455X (gfx1250) — matrix operand order repaired and checked by reading, not compiled, not run on hardware
//
#include <hip/hip_runtime.h>
#include <hip/hip_bf16.h>

typedef __bf16 bf16;
typedef __attribute__((ext_vector_type(8)))  __bf16 bf16x8;
typedef __attribute__((ext_vector_type(16))) __bf16 bf16x16;
typedef __attribute__((ext_vector_type(8)))  float  f32x8;

#define DEV __device__ __forceinline__

static constexpr int Bsz = 2, S = 2048, D = 1024, H = 16, DK = 64;

DEV bf16x16 cat8(bf16x8 lo, bf16x8 hi) {
  return __builtin_shufflevector(lo, hi, 0,1,2,3,4,5,6,7,8,9,10,11,12,13,14,15);
}
DEV bf16x16 cvt_cat(f32x8 lo, f32x8 hi) {
  return cat8(__builtin_convertvector(lo, bf16x8), __builtin_convertvector(hi, bf16x8));
}
DEV f32x8 wmma_bf16(bf16x16 a, bf16x16 b, f32x8 c) {
  // D = A(16x32 bf16) * B(32x16 bf16) + C(16x16 f32)
  return __builtin_amdgcn_wmma_f32_16x16x32_bf16(false, a, false, b, (short)0, c, false, false);
}

// lane-xor permute within 8-lane groups as a VALU op (v_mov_b32 dpp8)
constexpr int dpp8_sel_xor(int m) {
  int s = 0;
  for (int j = 0; j < 8; ++j) s |= ((j ^ m) & 7) << (3 * j);
  return s;
}
template <int M>
DEV float fmax_xor_dpp8(float v) {
  const int t = __builtin_amdgcn_mov_dpp8(__builtin_bit_cast(int, v), dpp8_sel_xor(M));
  return fmaxf(v, __builtin_bit_cast(float, t));
}

// ---------------------------------------------------------------------------
// Projection GEMM: Y[row,e] = (sum_d X[row,d] * W[e,d] + bias[e]) * sc
// One wave computes a 16x64 strip (4 N-tiles, A-fragment reused 4x).
//   MODE 0: bf16 out, layout [b,h,s,dk]        (Q with sc=0.125, K)
//   MODE 1: bf16 out, layout [b,h,dk,s] (V^T)  (V)
//   MODE 2: f32  out, layout [row, e] = d_out  (final O projection)
// ---------------------------------------------------------------------------
template <int MODE, typename XT>
__global__ __launch_bounds__(128) void proj_kernel(const XT* __restrict__ X,
                                                   const float* __restrict__ W,
                                                   const float* __restrict__ bias,
                                                   void* __restrict__ outp,
                                                   float sc) {
  const int lane = threadIdx.x & 31;
  const int wave = threadIdx.x >> 5;
  const int half = lane >> 4;   // which 16-lane group
  const int l15  = lane & 15;
  const int off1 = half * 8;    // A-fragment K offset for this half

  const int m0 = blockIdx.x * 16;                      // row tile over B*S
  const int n0 = (blockIdx.y * 16 + wave * 4) * 16;    // first of 4 col tiles

  const XT* xrow = X + (size_t)(m0 + l15) * D;         // A: lane&15 = row M
  const float* wrow[4];
#pragma unroll
  for (int t = 0; t < 4; ++t) wrow[t] = W + (size_t)(n0 + t * 16 + l15) * D;

  f32x8 c[4] = {};
#pragma unroll 2
  for (int kk = 0; kk < D / 32; ++kk) {
    const int ka = kk * 32 + off1;  // A: two 8-elem chunks (K and K+16)
    bf16x16 a;
    if constexpr (__is_same(XT, float)) {
      a = cvt_cat(*(const f32x8*)(xrow + ka), *(const f32x8*)(xrow + ka + 16));
    } else {
      a = cat8(*(const bf16x8*)(xrow + ka), *(const bf16x8*)(xrow + ka + 16));
    }
    const int kb = kk * 32 + off1;  // B: two 8-elem chunks (K and K+16)
#pragma unroll
    for (int t = 0; t < 4; ++t) {
      bf16x16 b = cvt_cat(*(const f32x8*)(wrow[t] + kb), *(const f32x8*)(wrow[t] + kb + 16));
      c[t] = wmma_bf16(a, b, c[t]);
    }
  }

#pragma unroll
  for (int t = 0; t < 4; ++t) {
    const int   e  = n0 + t * 16 + l15;  // output column for this lane
    const float bv = bias[e];

    if constexpr (MODE == 0) {
      bf16* out = (bf16*)outp;
      const int h = e >> 6, d = e & 63;
#pragma unroll
      for (int r = 0; r < 8; ++r) {
        const int row = m0 + r + 8 * half;  // global row over B*S
        const int b = row >> 11, s = row & (S - 1);
        out[(size_t)((b * H + h) * S + s) * DK + d] = (bf16)((c[t][r] + bv) * sc);
      }
    } else if constexpr (MODE == 1) {
      bf16* out = (bf16*)outp;
      const int h = e >> 6, d = e & 63;
      const int b = m0 >> 11;
      const int sbase = (m0 & (S - 1)) + 8 * half;  // rows contiguous in s
      bf16x8 pk;
#pragma unroll
      for (int r = 0; r < 8; ++r) pk[r] = (bf16)((c[t][r] + bv) * sc);
      *(bf16x8*)(out + (size_t)((b * H + h) * DK + d) * S + sbase) = pk;
    } else {
      float* out = (float*)outp;
#pragma unroll
      for (int r = 0; r < 8; ++r) {
        const int row = m0 + r + 8 * half;
        out[(size_t)row * D + e] = (c[t][r] + bv) * sc;
      }
    }
  }
}

// ---------------------------------------------------------------------------
// One 32-key flash-attention step (9 WMMAs: 4 QK^T, 1 row-sum, 4 PV).
// ---------------------------------------------------------------------------
template <bool MASKED>
DEV void attn_step(const bf16* __restrict__ kbase, const bf16* __restrict__ vbase,
                   bf16* __restrict__ pl, int j0, int q0,
                   int half, int l15, int off1,
                   const bf16x16& aq0, const bf16x16& aq1, const bf16x16& ones,
                   f32x8& o0, f32x8& o1, f32x8& o2, f32x8& o3,
                   float (&m_i)[8], float (&l_i)[8]) {
  // ---- scores = Q K^T for keys [j0, j0+32): two 16x16 C tiles ----
  const bf16* krow0 = kbase + (size_t)(j0 + l15) * DK;       // keys j0..+15
  const bf16* krow1 = kbase + (size_t)(j0 + 16 + l15) * DK;  // keys j0+16..+31
  f32x8 s0 = {}, s1 = {};
  s0 = wmma_bf16(aq0, *(const bf16x16*)(krow0 + half * 16),      s0);
  s0 = wmma_bf16(aq1, *(const bf16x16*)(krow0 + 32 + half * 16), s0);
  s1 = wmma_bf16(aq0, *(const bf16x16*)(krow1 + half * 16),      s1);
  s1 = wmma_bf16(aq1, *(const bf16x16*)(krow1 + 32 + half * 16), s1);

  float p0[8], p1[8];
#pragma unroll
  for (int r = 0; r < 8; ++r) {
    float v0 = s0[r], v1 = s1[r];
    if constexpr (MASKED) {
      const int mrow = q0 + r + 8 * half;
      if (j0 + l15      > mrow) v0 = -1.0e30f;
      if (j0 + 16 + l15 > mrow) v1 = -1.0e30f;
    }
    p0[r] = v0; p1[r] = v1;
  }

  // ---- running max: xor-1/2/4 via DPP8 (VALU), xor-8 via one bpermute ----
  float alpha[8];
#pragma unroll
  for (int r = 0; r < 8; ++r) {
    float v = fmaxf(p0[r], p1[r]);
    v = fmax_xor_dpp8<1>(v);
    v = fmax_xor_dpp8<2>(v);
    v = fmax_xor_dpp8<4>(v);
    v = fmaxf(v, __shfl_xor(v, 8));
    const float mn = fmaxf(m_i[r], v);
    alpha[r] = __expf(m_i[r] - mn);
    m_i[r]   = mn;
    p0[r] = __expf(p0[r] - mn);
    p1[r] = __expf(p1[r] - mn);
  }

  // ---- C-layout -> A-layout via LDS (bf16) ----
#pragma unroll
  for (int r = 0; r < 8; ++r) {
    const int row = r + 8 * half;
    pl[row * 32 + l15]      = (bf16)p0[r];
    pl[row * 32 + 16 + l15] = (bf16)p1[r];
  }
  asm volatile("s_wait_dscnt 0" ::: "memory");  // DS in-order per wave; order compiler too
  const bf16x16 ap = cat8(*(const bf16x8*)(pl + l15 * 32 + half * 16),
                          *(const bf16x8*)(pl + l15 * 32 + half * 16 + 8));

  // ---- row sums via WMMA against all-ones B (no cross-lane reduction) ----
  f32x8 rs = {};
  rs = wmma_bf16(ap, ones, rs);
#pragma unroll
  for (int r = 0; r < 8; ++r) {
    l_i[r] = l_i[r] * alpha[r] + rs[r];
    o0[r] *= alpha[r]; o1[r] *= alpha[r]; o2[r] *= alpha[r]; o3[r] *= alpha[r];
  }

  // ---- O += P * V : V^T gives contiguous 32B B-fragments ----
  const bf16* vcol = vbase + (size_t)l15 * S + j0 + half * 16;
  o0 = wmma_bf16(ap, *(const bf16x16*)(vcol),          o0);
  o1 = wmma_bf16(ap, *(const bf16x16*)(vcol + 16 * S), o1);
  o2 = wmma_bf16(ap, *(const bf16x16*)(vcol + 32 * S), o2);
  o3 = wmma_bf16(ap, *(const bf16x16*)(vcol + 48 * S), o3);
}

// ---------------------------------------------------------------------------
// Causal flash attention. One wave per 16-row q-tile; 32 keys per step.
// Q,K: bf16 [b,h,s,64] (Q pre-scaled by 0.125); V: bf16 [b,h,64,s]; O: [b,s,D].
// Only the final 32-key block intersects the diagonal -> maskless main loop.
// ---------------------------------------------------------------------------
__global__ __launch_bounds__(128) void attn_kernel(const bf16* __restrict__ Qb,
                                                   const bf16* __restrict__ Kb,
                                                   const bf16* __restrict__ Vt,
                                                   bf16* __restrict__ Ob) {
  __shared__ bf16 Plds[4][16 * 32];  // per-wave P tile, row-major 16x32

  const int lane = threadIdx.x & 31;
  const int wave = threadIdx.x >> 5;
  const int half = lane >> 4;
  const int l15  = lane & 15;
  const int off1 = half * 8;

  const int qt = blockIdx.x * 4 + wave;
  const int h  = blockIdx.y;
  const int b  = blockIdx.z;
  const int q0 = qt * 16;

  const bf16* kbase = Kb + (size_t)(b * H + h) * S * DK;
  const bf16* vbase = Vt + (size_t)(b * H + h) * DK * S;

  // Q A-fragments for d-chunks [0,32) and [32,64)
  const bf16* qrow = Qb + (size_t)(b * H + h) * S * DK + (size_t)(q0 + l15) * DK;
  const bf16x16 aq0 = cat8(*(const bf16x8*)(qrow + half * 16),
                           *(const bf16x8*)(qrow + half * 16 + 8));
  const bf16x16 aq1 = cat8(*(const bf16x8*)(qrow + 32 + half * 16),
                           *(const bf16x8*)(qrow + 32 + half * 16 + 8));

  bf16x16 ones;
#pragma unroll
  for (int i = 0; i < 16; ++i) ones[i] = (bf16)1.0f;

  f32x8 o0 = {}, o1 = {}, o2 = {}, o3 = {};
  float m_i[8], l_i[8];
#pragma unroll
  for (int r = 0; r < 8; ++r) { m_i[r] = -3.0e38f; l_i[r] = 0.0f; }

  bf16* pl = Plds[wave];

  // blocks with j0+31 <= q0 need no mask; exactly one masked block follows
  const int nfull = (q0 + 1) / 32;
  for (int jt = 0; jt < nfull; ++jt)
    attn_step<false>(kbase, vbase, pl, jt * 32, q0, half, l15, off1,
                     aq0, aq1, ones, o0, o1, o2, o3, m_i, l_i);
  attn_step<true>(kbase, vbase, pl, nfull * 32, q0, half, l15, off1,
                  aq0, aq1, ones, o0, o1, o2, o3, m_i, l_i);

  // ---- normalize and store O in [b,s,D] (head-major concat) ----
#pragma unroll
  for (int r = 0; r < 8; ++r) {
    const float inv = 1.0f / l_i[r];
    const int s = q0 + r + 8 * half;
    bf16* orow = Ob + (size_t)(b * S + s) * D + h * DK;
    orow[l15]      = (bf16)(o0[r] * inv);
    orow[16 + l15] = (bf16)(o1[r] * inv);
    orow[32 + l15] = (bf16)(o2[r] * inv);
    orow[48 + l15] = (bf16)(o3[r] * inv);
  }
}

// ---------------------------------------------------------------------------
extern "C" void kernel_launch(void* const* d_in, const int* in_sizes, int n_in,
                              void* d_out, int out_size, void* d_ws, size_t ws_size,
                              hipStream_t stream) {
  (void)in_sizes; (void)n_in; (void)out_size; (void)ws_size;

  const float* q_in = (const float*)d_in[0];
  const float* k_in = (const float*)d_in[1];
  const float* v_in = (const float*)d_in[2];
  const float* w_q  = (const float*)d_in[3];
  const float* b_q  = (const float*)d_in[4];
  const float* w_k  = (const float*)d_in[5];
  const float* b_k  = (const float*)d_in[6];
  const float* w_v  = (const float*)d_in[7];
  const float* b_v  = (const float*)d_in[8];
  const float* w_o  = (const float*)d_in[9];
  const float* b_o  = (const float*)d_in[10];

  const size_t NE = (size_t)Bsz * S * D;  // 4M elements per tensor
  bf16* Qb = (bf16*)d_ws;
  bf16* Kb = Qb + NE;
  bf16* Vt = Kb + NE;
  bf16* Ob = Vt + NE;

  const dim3 pblk(128);
  const dim3 pgrd(Bsz * S / 16, D / 256);  // 256 x 4 blocks; 16x64 strip per wave

  // Fold 1/sqrt(dk) into Q so the attention kernel skips the score scaling.
  proj_kernel<0, float><<<pgrd, pblk, 0, stream>>>(q_in, w_q, b_q, Qb, 0.125f);
  proj_kernel<0, float><<<pgrd, pblk, 0, stream>>>(k_in, w_k, b_k, Kb, 1.0f);
  proj_kernel<1, float><<<pgrd, pblk, 0, stream>>>(v_in, w_v, b_v, Vt, 1.0f);

  attn_kernel<<<dim3(S / 64, H, Bsz), 128, 0, stream>>>(Qb, Kb, Vt, Ob);

  proj_kernel<2, bf16><<<pgrd, pblk, 0, stream>>>(Ob, w_o, b_o, d_out, 1.0f);
}
